// CGCNNEncoder_86586540687375
// MI455X (gfx1250) — compile-verified, matrix operand order checked by reading, not run on hardware
//
#include <hip/hip_runtime.h>
#include <hip/hip_bf16.h>

// ---------------------------------------------------------------------------
// CGCNN-style 3-layer GCN encoder for MI455X (gfx1250, wave32, WMMA).
//  * GEMMs: v_wmma_f32_16x16x32_f16, 64-row strips per block, B fragments
//    held in VGPRs across 4 row tiles, A staged to LDS via async-to-LDS
//    (ASYNCcnt). Tail handling is block-uniform so the hot path has no
//    per-lane guards.
//  * Edge aggregation: float4 gathers + global f32 atomics (L2-resident).
//  * BatchNorm: LDS block reductions + 128-address atomics.
// ---------------------------------------------------------------------------

#define DH    128          // node/hidden dim
#define GROWS 64           // GEMM rows per block
#define BNEPS 1e-5f

typedef __attribute__((ext_vector_type(16))) _Float16 v16h;
typedef __attribute__((ext_vector_type(8)))  _Float16 v8h;
typedef __attribute__((ext_vector_type(4)))  _Float16 v4h;
typedef __attribute__((ext_vector_type(8)))  float    v8f;
typedef __attribute__((ext_vector_type(4)))  int      v4i;
// Address-space qualified int4, matching the async-to-LDS builtin prototype.
typedef __attribute__((address_space(1))) v4i as1_v4i;
typedef __attribute__((address_space(3))) v4i as3_v4i;

// ---------------------------------------------------------------- utilities
__global__ void k_zero_f32(float* __restrict__ p, long n) {
  long i = (long)blockIdx.x * blockDim.x + threadIdx.x;
  if (i < n) p[i] = 0.0f;
}

__global__ void k_degree(const int* __restrict__ ei, float* __restrict__ deg, int E) {
  long e = (long)blockIdx.x * blockDim.x + threadIdx.x;
  if (e < E) atomicAdd(&deg[ei[E + e]], 1.0f);
}

__global__ void k_deg_inv(float* __restrict__ d, int n) {
  long i = (long)blockIdx.x * blockDim.x + threadIdx.x;
  if (i < n) d[i] = rsqrtf(d[i] + 1.0f);
}

// Wt[col*128 + k] = (f16) W[k*128 + col]   (column-major f16 weight)
__global__ void k_w_to_f16t(const float* __restrict__ W, _Float16* __restrict__ Wt) {
  int t = blockIdx.x * blockDim.x + threadIdx.x;   // 16384 threads
  int col = t >> 7, k = t & 127;
  Wt[(col << 7) + k] = (_Float16)W[(k << 7) + col];
}

// f32 -> f16, vectorized (used once for the input features x)
__global__ void k_cast_f16(const float* __restrict__ src, _Float16* __restrict__ dst, long n4) {
  long i = (long)blockIdx.x * blockDim.x + threadIdx.x;
  if (i >= n4) return;
  float4 v = *(const float4*)(src + i * 4);
  v4h o;
  o.x = (_Float16)v.x; o.y = (_Float16)v.y;
  o.z = (_Float16)v.z; o.w = (_Float16)v.w;
  *(v4h*)(dst + i * 4) = o;
}

// --------------------------------------------------------------- WMMA GEMM
// Out[N,128] = Ah[N,128](f16) @ W[128,128]; W as f16 col-major Wt.
// Block = 256 threads = 8 waves; block owns a 64-row strip; wave w owns the
// 16-col tile [16w,16w+16) and reuses its 4 B fragments across 4 row tiles.
__global__ __launch_bounds__(256) void k_gemm_wmma(
    const _Float16* __restrict__ Ah, const _Float16* __restrict__ Wt,
    float* __restrict__ Out, int n) {
  __shared__ __align__(16) _Float16 As[GROWS * DH];
  const int row0 = blockIdx.x * GROWS;
  const int t = threadIdx.x;
  const bool full = (row0 + GROWS) <= n;   // block-uniform tail predicate

  // ---- Stage 64x128 f16 A strip into LDS: 1024 x b128, 4 per thread.
  if (full) {
#pragma unroll
    for (int j = 0; j < 4; ++j) {
      int eidx = (t + 256 * j) * 8;         // half-element index within strip
      const _Float16* gp = Ah + (size_t)row0 * DH + eidx;
      _Float16* lp = &As[eidx];
#if __has_builtin(__builtin_amdgcn_global_load_async_to_lds_b128)
      __builtin_amdgcn_global_load_async_to_lds_b128(
          (as1_v4i*)(v4i*)const_cast<_Float16*>(gp), (as3_v4i*)(v4i*)lp, 0, 0);
#else
      *(v8h*)lp = *(const v8h*)gp;
#endif
    }
  } else {
#pragma unroll
    for (int j = 0; j < 4; ++j) {
      int eidx = (t + 256 * j) * 8;
      int r = eidx >> 7;
      int c = eidx & 127;
      int gr = row0 + r; if (gr >= n) gr = n - 1;    // clamp tail rows
      const _Float16* gp = Ah + (size_t)gr * DH + c;
      _Float16* lp = &As[r * DH + c];
#if __has_builtin(__builtin_amdgcn_global_load_async_to_lds_b128)
      __builtin_amdgcn_global_load_async_to_lds_b128(
          (as1_v4i*)(v4i*)const_cast<_Float16*>(gp), (as3_v4i*)(v4i*)lp, 0, 0);
#else
      *(v8h*)lp = *(const v8h*)gp;
#endif
    }
  }
#if __has_builtin(__builtin_amdgcn_global_load_async_to_lds_b128)
#if __has_builtin(__builtin_amdgcn_s_wait_asynccnt)
  __builtin_amdgcn_s_wait_asynccnt(0);
#else
  asm volatile("s_wait_asynccnt 0" ::: "memory");
#endif
#endif
  // Speculative prefetch of the next strip (global_prefetch_b8).
  if (row0 + GROWS < n)
    __builtin_prefetch(Ah + (size_t)(row0 + GROWS) * DH + t * 8, 0, 0);
  __syncthreads();

  const int wave = t >> 5;             // 0..7 -> 16-col tile
  const int lane = t & 31;
  const int colbase = wave * 16;
  const int m16 = lane & 15;           // row within A tile / col within B tile
  const int kb  = (lane >> 4) * 8;     // K sub-offset per lane half

  // ---- Preload B fragments for all 4 K chunks (reused by 4 row tiles).
  union Frag { v16h v; v8h h[2]; };
  Frag bfr[4];
#pragma unroll
  for (int kcI = 0; kcI < 4; ++kcI) {
    const _Float16* wp = Wt + (size_t)(colbase + m16) * DH + kcI * 32 + kb;
    bfr[kcI].h[0] = *(const v8h*)wp;
    bfr[kcI].h[1] = *(const v8h*)(wp + 16);
  }

  // ---- 4 row tiles x 4 K chunks = 16 WMMAs per wave.
#pragma unroll
  for (int rt = 0; rt < 4; ++rt) {
    v8f acc = {};
    const int ar = rt * 16 + m16;
#pragma unroll
    for (int kcI = 0; kcI < 4; ++kcI) {
      Frag a;
      a.h[0] = *(const v8h*)&As[ar * DH + kcI * 32 + kb];
      a.h[1] = *(const v8h*)&As[ar * DH + kcI * 32 + kb + 16];
      acc = __builtin_amdgcn_wmma_f32_16x16x32_f16(
          false, a.v, false, bfr[kcI].v, (short)0, acc, false, false);
    }
    // C/D layout: VGPR r -> row r + 8*(lane>>4), col = colbase + (lane&15)
    const int rbase = row0 + rt * 16 + ((lane >> 4) << 3);
    float* p = Out + (size_t)rbase * DH + colbase + m16;
    if (full) {
      // fast path: one address, 8 stores at immediate offsets (r * 512 B)
#pragma unroll
      for (int r = 0; r < 8; ++r) p[r * DH] = acc[r];
    } else {
#pragma unroll
      for (int r = 0; r < 8; ++r)
        if (rbase + r < n) p[r * DH] = acc[r];
    }
  }
}

// --------------------------------------------------- GCN aggregation kernels
// agg[i,:] = hlin[i,:] * dinv[i]^2 + bias   (self-loop + bias init)
__global__ void k_init_agg(const float* __restrict__ hlin, const float* __restrict__ dinv,
                           const float* __restrict__ bias, float* __restrict__ agg, int n) {
  long tid = (long)blockIdx.x * blockDim.x + threadIdx.x;
  long i = tid >> 5;
  if (i >= n) return;
  int c = (int)(tid & 31) * 4;
  float sn = dinv[i]; sn *= sn;
  float4 v = *(const float4*)(hlin + i * DH + c);
  float4 o;
  o.x = v.x * sn + bias[c + 0];
  o.y = v.y * sn + bias[c + 1];
  o.z = v.z * sn + bias[c + 2];
  o.w = v.w * sn + bias[c + 3];
  *(float4*)(agg + i * DH + c) = o;
}

// agg[dst,:] += hlin[src,:] * dinv[src]*dinv[dst]   (32 threads / edge)
__global__ void k_edge_agg(const int* __restrict__ ei, const float* __restrict__ hlin,
                           const float* __restrict__ dinv, float* __restrict__ agg, int E) {
  long tid = (long)blockIdx.x * blockDim.x + threadIdx.x;
  long e = tid >> 5;
  if (e >= E) return;
  int c = (int)(tid & 31) * 4;
  int src = ei[e];
  int dst = ei[E + e];
  float nrm = dinv[src] * dinv[dst];
  float4 v = *(const float4*)(hlin + (size_t)src * DH + c);
  float* p = agg + (size_t)dst * DH + c;
  atomicAdd(p + 0, v.x * nrm);
  atomicAdd(p + 1, v.y * nrm);
  atomicAdd(p + 2, v.z * nrm);
  atomicAdd(p + 3, v.w * nrm);
}

// ----------------------------------------------------------- BatchNorm path
__global__ __launch_bounds__(256) void k_bn_stats(
    const float* __restrict__ h, int n,
    float* __restrict__ gsum, float* __restrict__ gsq) {
  __shared__ float ssum[256], ssq[256];
  int t = threadIdx.x;
  int c = t & 127;
  int half = t >> 7;
  float s = 0.f, q = 0.f;
  for (long r = (long)blockIdx.x * 2 + half; r < n; r += (long)gridDim.x * 2) {
    float v = h[r * DH + c];
    s += v; q += v * v;
  }
  ssum[t] = s; ssq[t] = q;
  __syncthreads();
  if (t < 128) {
    atomicAdd(&gsum[c], ssum[t] + ssum[t + 128]);
    atomicAdd(&gsq[c],  ssq[t]  + ssq[t + 128]);
  }
}

__global__ void k_bn_final(const float* __restrict__ gsum, const float* __restrict__ gsq,
                           const float* __restrict__ g, const float* __restrict__ be,
                           float* __restrict__ scale, float* __restrict__ shift, int n) {
  int c = threadIdx.x;   // 128 threads
  float invn = 1.0f / (float)n;
  float m = gsum[c] * invn;
  float v = gsq[c] * invn - m * m;
  float sc = g[c] * rsqrtf(v + BNEPS);
  scale[c] = sc;
  shift[c] = be[c] - m * sc;
}

// h = sigmoid(h*scale[c]+shift[c]) in place, plus f16 copy for the next GEMM.
__global__ void k_bn_apply(float* __restrict__ h, const float* __restrict__ scale,
                           const float* __restrict__ shift, _Float16* __restrict__ hf, int n) {
  long tid = (long)blockIdx.x * blockDim.x + threadIdx.x;
  long i = tid >> 5;
  if (i >= n) return;
  int c = (int)(tid & 31) * 4;
  float4 v = *(const float4*)(h + i * DH + c);
  float4 o;
  o.x = 1.0f / (1.0f + __expf(-(v.x * scale[c + 0] + shift[c + 0])));
  o.y = 1.0f / (1.0f + __expf(-(v.y * scale[c + 1] + shift[c + 1])));
  o.z = 1.0f / (1.0f + __expf(-(v.z * scale[c + 2] + shift[c + 2])));
  o.w = 1.0f / (1.0f + __expf(-(v.w * scale[c + 3] + shift[c + 3])));
  *(float4*)(h + i * DH + c) = o;
  v4h of;
  of.x = (_Float16)o.x; of.y = (_Float16)o.y;
  of.z = (_Float16)o.z; of.w = (_Float16)o.w;
  *(v4h*)(hf + i * DH + c) = of;
}

// --------------------------------------------------------------- mean pool
__global__ void k_pool_acc(const float* __restrict__ h, const int* __restrict__ batch,
                           float* __restrict__ out, float* __restrict__ cnt, int n) {
  long tid = (long)blockIdx.x * blockDim.x + threadIdx.x;
  long i = tid >> 5;
  if (i >= n) return;
  int c = (int)(tid & 31) * 4;
  int g = batch[i];
  float4 v = *(const float4*)(h + i * DH + c);
  float* p = out + (size_t)g * DH + c;
  atomicAdd(p + 0, v.x);
  atomicAdd(p + 1, v.y);
  atomicAdd(p + 2, v.z);
  atomicAdd(p + 3, v.w);
  if ((tid & 31) == 0) atomicAdd(&cnt[g], 1.0f);
}

__global__ void k_pool_div(float* __restrict__ out, const float* __restrict__ cnt, int total) {
  int t = blockIdx.x * blockDim.x + threadIdx.x;
  if (t < total) out[t] /= fmaxf(cnt[t >> 7], 1.0f);
}

// ------------------------------------------------------------------- driver
static inline long cdiv(long a, long b) { return (a + b - 1) / b; }

extern "C" void kernel_launch(void* const* d_in, const int* in_sizes, int n_in,
                              void* d_out, int out_size, void* d_ws, size_t ws_size,
                              hipStream_t stream) {
  (void)n_in; (void)ws_size;
  const float* x     = (const float*)d_in[0];
  const int*   ei    = (const int*)d_in[1];
  const int*   batch = (const int*)d_in[3];
  const float* W[3]  = {(const float*)d_in[4], (const float*)d_in[6], (const float*)d_in[8]};
  const float* bi[3] = {(const float*)d_in[5], (const float*)d_in[7], (const float*)d_in[9]};
  const float* ga[3] = {(const float*)d_in[10], (const float*)d_in[12], (const float*)d_in[14]};
  const float* be[3] = {(const float*)d_in[11], (const float*)d_in[13], (const float*)d_in[15]};

  const int n = in_sizes[0] / DH;     // 50000 nodes
  const int E = in_sizes[1] / 2;      // 800000 edges
  const int G = out_size / DH;        // 64 graphs

  // ---- workspace carve-up (256B aligned blocks)
  auto aup = [](size_t v) { return (v + 255) & ~size_t(255); };
  char* w = (char*)d_ws;
  float*    dinv = (float*)w;            w += aup((size_t)n * 4);
  float*    bufL = (float*)w;            w += aup((size_t)n * DH * 4);   // h @ W
  float*    buf0 = (float*)w;            w += aup((size_t)n * DH * 4);   // agg / h (ping)
  float*    buf1 = (float*)w;            w += aup((size_t)n * DH * 4);   // agg / h (pong)
  _Float16* hf16 = (_Float16*)w;         w += aup((size_t)n * DH * 2);   // f16 activations
  _Float16* Wt   = (_Float16*)w;         w += aup((size_t)DH * DH * 2);  // f16 W^T
  float*    gsum = (float*)w;            w += aup(DH * 4);
  float*    gsq  = (float*)w;            w += aup(DH * 4);
  float*    scl  = (float*)w;            w += aup(DH * 4);
  float*    shf  = (float*)w;            w += aup(DH * 4);
  float*    cnt  = (float*)w;            w += aup((size_t)G * 4);

  const int TB = 256;
  const long nElem4 = (long)n * 32;     // (node, 4-channel) work items
  const long eElem4 = (long)E * 32;     // (edge, 4-channel) work items

  // ---- degrees -> dinv
  k_zero_f32<<<cdiv(n, TB), TB, 0, stream>>>(dinv, n);
  k_degree  <<<cdiv(E, TB), TB, 0, stream>>>(ei, dinv, E);
  k_deg_inv <<<cdiv(n, TB), TB, 0, stream>>>(dinv, n);

  // ---- input features to f16 once
  k_cast_f16<<<cdiv((long)n * DH / 4, TB), TB, 0, stream>>>(x, hf16, (long)n * DH / 4);

  float* hbuf[2] = {buf0, buf1};
  const float* hin_f32 = nullptr;       // f32 activations of the latest layer
  for (int layer = 0; layer < 3; ++layer) {
    float* agg = hbuf[layer & 1];       // becomes the layer's output h (in place)

    k_w_to_f16t<<<(DH * DH) / TB, TB, 0, stream>>>(W[layer], Wt);
    k_gemm_wmma<<<cdiv(n, GROWS), TB, 0, stream>>>(hf16, Wt, bufL, n);
    k_init_agg <<<cdiv(nElem4, TB), TB, 0, stream>>>(bufL, dinv, bi[layer], agg, n);
    k_edge_agg <<<cdiv(eElem4, TB), TB, 0, stream>>>(ei, bufL, dinv, agg, E);

    k_zero_f32 <<<1, TB, 0, stream>>>(gsum, DH);
    k_zero_f32 <<<1, TB, 0, stream>>>(gsq, DH);
    k_bn_stats <<<200, TB, 0, stream>>>(agg, n, gsum, gsq);
    k_bn_final <<<1, DH, 0, stream>>>(gsum, gsq, ga[layer], be[layer], scl, shf, n);
    k_bn_apply <<<cdiv(nElem4, TB), TB, 0, stream>>>(agg, scl, shf, hf16, n);

    hin_f32 = agg;
  }

  // ---- global mean pool into d_out
  float* out = (float*)d_out;
  k_zero_f32 <<<cdiv((long)G * DH, TB), TB, 0, stream>>>(out, (long)G * DH);
  k_zero_f32 <<<1, TB, 0, stream>>>(cnt, G);
  k_pool_acc <<<cdiv(nElem4, TB), TB, 0, stream>>>(hin_f32, batch, out, cnt, n);
  k_pool_div <<<cdiv((long)G * DH, TB), TB, 0, stream>>>(out, cnt, G * DH);
}
